// CLF_block_28226525069391
// MI455X (gfx1250) — matrix operand order repaired and checked by reading, not compiled, not run on hardware
//
#include <hip/hip_runtime.h>

// ---------- types ----------
typedef __attribute__((ext_vector_type(16))) __bf16 v16bf;
typedef __attribute__((ext_vector_type(8)))  float  v8f;

#define HWN 16384   // H*W
#define CCH 256     // channels

// LDS tile geometry: rows of 64B (32 bf16) with 16B pad every 4 rows.
// row byte offset = r*64 + (r>>2)*16  -> 16B aligned (async B128) and
// fragment reads hit distinct banks across 16 lanes.
#define T128US 4352          // ushorts in a 128-row tile (8704 B)
#define T64US  2176          // ushorts in a  64-row tile (4352 B)

__device__ __forceinline__ int row_us(int r) { return (r << 5) + ((r >> 2) << 3); }
__device__ __forceinline__ unsigned row_b(int r) {
  return (unsigned)((r << 6) + ((r >> 2) << 4));
}

union Frag16 {
  v16bf v;
  unsigned int u[8];
};

__device__ __forceinline__ unsigned short f32_to_bf16(float f) {
  unsigned int x = __builtin_bit_cast(unsigned int, f);
  x += 0x7fffu + ((x >> 16) & 1u);          // round-to-nearest-even
  return (unsigned short)(x >> 16);
}

// A operand (16x32 bf16): M in lanes, K-pairs per VGPR per ISA layout
__device__ __forceinline__ void load_frag_a(Frag16& fr, const unsigned short* p,
                                            int row0, int lane) {
  int hf = lane >> 4;
  const unsigned short* q = p + row_us(row0 + (lane & 15));
#pragma unroll
  for (int v = 0; v < 8; ++v) {
    int ko = ((v & 4) ? 16 : 0) + 8 * hf + 2 * (v & 3);
    fr.u[v] = *(const unsigned int*)&q[ko];
  }
}

// B operand (32x16 bf16): N in lanes, K-pairs per VGPR
__device__ __forceinline__ void load_frag_b(Frag16& fr, const unsigned short* p,
                                            int row0, int lane) {
  int hf = lane >> 4;
  const unsigned short* q = p + row_us(row0 + (lane & 15));
#pragma unroll
  for (int v = 0; v < 8; ++v)
    fr.u[v] = *(const unsigned int*)&q[((v & 4) ? 16 : 0) + 8 * hf + 2 * (v & 3)];
}

__device__ __forceinline__ v8f wmma_bf16(const Frag16& A, const Frag16& B, v8f C) {
  return __builtin_amdgcn_wmma_f32_16x16x32_bf16(false, A.v, false, B.v,
                                                 (short)0, C, false, false);
}

__device__ __forceinline__ v8f zero8() {
  v8f z;
#pragma unroll
  for (int e = 0; e < 8; ++e) z[e] = 0.0f;
  return z;
}

// ---------- CDNA5 async global->LDS DMA (ASYNCcnt-tracked) ----------
__device__ __forceinline__ void async_b128(unsigned lds_byte_addr,
                                           const void* sbase, unsigned voff) {
  asm volatile("global_load_async_to_lds_b128 %0, %1, %2"
               :: "v"(lds_byte_addr), "v"(voff), "s"(sbase));
}
__device__ __forceinline__ void wait_async0() {
  asm volatile("s_wait_asynccnt 0" ::: "memory");
}
__device__ __forceinline__ unsigned lds_addr_of(const void* p) {
  // LDS aperture: flat address low 32 bits are the LDS byte offset
  return (unsigned)(unsigned long long)p;
}

// ---------- kernel 1: compose weights  Wt = w_t @ w1 (bf16),  bias_t = w_t@b1 + b_t ----------
__global__ __launch_bounds__(256) void compose_kernel(
    const float* __restrict__ w1, const float* __restrict__ b1,
    const float* __restrict__ w2, const float* __restrict__ b2,
    const float* __restrict__ w3, const float* __restrict__ b3,
    const float* __restrict__ w4, const float* __restrict__ b4,
    unsigned short* __restrict__ Wbf, float* __restrict__ biasf) {
  int mat = blockIdx.x >> 8;
  int o   = blockIdx.x & 255;
  int c   = threadIdx.x;
  const float* wsrc = (mat == 0) ? w2 : (mat == 1) ? w3 : w4;
  const float* bsrc = (mat == 0) ? b2 : (mat == 1) ? b3 : b4;
  float acc = 0.0f;
  for (int j = 0; j < CCH; ++j)
    acc += wsrc[o * CCH + j] * w1[j * CCH + c];
  Wbf[mat * 65536 + o * CCH + c] = f32_to_bf16(acc);
  if (c == 0) {
    float bb = bsrc[o];
    for (int j = 0; j < CCH; ++j) bb += wsrc[o * CCH + j] * b1[j];
    biasf[mat * CCH + o] = bb;
  }
}

// ---------- kernel 2: q,k,v = Wc @ concat(a,b) + bias ----------
// block tile: 128 cout x 64 n, 8 waves; W tiles via async DMA, x via VALU convert
__global__ __launch_bounds__(256) void qkv_kernel(
    const float* __restrict__ a, const float* __restrict__ b,
    const unsigned short* __restrict__ Wbf, const float* __restrict__ biasf,
    unsigned short* __restrict__ qg, unsigned short* __restrict__ kg,
    unsigned short* __restrict__ vtg) {
  __shared__ __align__(16) unsigned short lw[3 * T128US];  // W tiles [mat][co][k]
  __shared__ __align__(16) unsigned short lx[T64US];       // x tile transposed [n][c]
  __shared__ unsigned short lvt[64 * 130];                 // v transpose staging

  int tid  = threadIdx.x;
  int lane = tid & 31, wid = tid >> 5;
  int n0  = blockIdx.x * 64;
  int co0 = blockIdx.y * 128;
  int bat = blockIdx.z;
  int m0  = wid * 16;

  // per-thread async chunk mapping for the 3 weight tiles (3*512 chunks of 16B)
  unsigned w_lds[6], w_gl[6];
  unsigned lwb = lds_addr_of(&lw[0]);
#pragma unroll
  for (int j = 0; j < 6; ++j) {
    int ch = tid + j * 256;
    int mat = ch >> 9, rem = ch & 511;
    int r = rem >> 2, qo = rem & 3;
    w_lds[j] = lwb + (unsigned)(mat * (T128US * 2)) + row_b(r) + qo * 16;
    w_gl[j]  = (unsigned)(mat * 131072 + (co0 + r) * 512 + qo * 16);
  }

  v8f accq[4], acck[4], accv[4];
#pragma unroll
  for (int j = 0; j < 4; ++j) { accq[j] = zero8(); acck[j] = zero8(); accv[j] = zero8(); }

  for (int ks = 0; ks < 8; ++ks) {
    int kk0 = ks * 32;
    __syncthreads();
    // async DMA the three 128x32 bf16 weight tiles
#pragma unroll
    for (int j = 0; j < 6; ++j)
      async_b128(w_lds[j], Wbf, w_gl[j] + (unsigned)(kk0 * 2));
    // stage x tile: 32 channels x 64 n, fp32->bf16, transposed into [n][c]
    const float* src = (kk0 < 128) ? (a + (bat * 128 + kk0) * HWN)
                                   : (b + (bat * 128 + (kk0 - 128)) * HWN);
    for (int i = tid; i < 32 * 64; i += 256) {
      int kc = i >> 6, n = i & 63;
      lx[row_us(n) + kc] = f32_to_bf16(src[kc * HWN + n0 + n]);
    }
    wait_async0();
    __syncthreads();

    Frag16 Aq, Ak, Av;
    load_frag_a(Aq, lw,              m0, lane);
    load_frag_a(Ak, lw + T128US,     m0, lane);
    load_frag_a(Av, lw + 2 * T128US, m0, lane);
#pragma unroll
    for (int j = 0; j < 4; ++j) {
      Frag16 Bx;
      load_frag_b(Bx, lx, j * 16, lane);
      accq[j] = wmma_bf16(Aq, Bx, accq[j]);
      acck[j] = wmma_bf16(Ak, Bx, acck[j]);
      accv[j] = wmma_bf16(Av, Bx, accv[j]);
    }
  }

  int lane15 = lane & 15, hf = lane >> 4;
#pragma unroll
  for (int r = 0; r < 8; ++r) {
    int crel = m0 + r + 8 * hf;
    int c    = co0 + crel;
    float bq = biasf[c];
    float bk = biasf[CCH + c];
    float bv = biasf[2 * CCH + c];
#pragma unroll
    for (int j = 0; j < 4; ++j) {
      int n    = n0 + j * 16 + lane15;
      int base = (bat * CCH + c) * HWN + n;
      qg[base] = f32_to_bf16(accq[j][r] + bq);
      kg[base] = f32_to_bf16(acck[j][r] + bk);
      lvt[(j * 16 + lane15) * 130 + crel] = f32_to_bf16(accv[j][r] + bv);
    }
  }
  __syncthreads();
  for (int i = tid; i < 64 * 64; i += 256) {
    int n = i >> 6, cw = i & 63;
    unsigned int uval = *(const unsigned int*)&lvt[n * 130 + 2 * cw];
    *(unsigned int*)&vtg[(bat * HWN + n0 + n) * CCH + co0 + 2 * cw] = uval;
  }
}

// ---------- kernel 3: qk partials, 128x64 tile, async double-buffered ----------
__global__ __launch_bounds__(128) void qk_kernel(
    const unsigned short* __restrict__ qg, const unsigned short* __restrict__ kg,
    float* __restrict__ qkp) {
  __shared__ __align__(16) unsigned short sm[2][T128US + T64US];
  int tid = threadIdx.x, lane = tid & 31, wid = tid >> 5;
  int c0 = (blockIdx.x >> 2) * 128;
  int d0 = (blockIdx.x & 3) * 64;
  int ksp = blockIdx.y;
  int bat = blockIdx.z;
  int m0  = wid * 32;

  unsigned smb = lds_addr_of(&sm[0][0]);
  const unsigned bufB = (T128US + T64US) * 2;
  // q tile: 128 rows -> 512 chunks; k tile: 64 rows -> 256 chunks
  unsigned q_l[4], q_g[4], k_l[2], k_g[2];
#pragma unroll
  for (int j = 0; j < 4; ++j) {
    int ch = tid + j * 128, r = ch >> 2, qo = ch & 3;
    q_l[j] = smb + row_b(r) + qo * 16;
    q_g[j] = (unsigned)((bat * CCH + c0 + r) * HWN) * 2u + qo * 16;
  }
#pragma unroll
  for (int j = 0; j < 2; ++j) {
    int ch = tid + j * 128, r = ch >> 2, qo = ch & 3;
    k_l[j] = smb + T128US * 2 + row_b(r) + qo * 16;
    k_g[j] = (unsigned)((bat * CCH + d0 + r) * HWN) * 2u + qo * 16;
  }

  v8f acc0[4], acc1[4];
#pragma unroll
  for (int j = 0; j < 4; ++j) { acc0[j] = zero8(); acc1[j] = zero8(); }

  int nstart = ksp * 4096;
  {
    unsigned nn2 = (unsigned)nstart * 2u;
#pragma unroll
    for (int j = 0; j < 4; ++j) async_b128(q_l[j], qg, q_g[j] + nn2);
#pragma unroll
    for (int j = 0; j < 2; ++j) async_b128(k_l[j], kg, k_g[j] + nn2);
  }
  wait_async0();
  __syncthreads();

  for (int t = 0; t < 128; ++t) {
    int cur = t & 1;
    if (t + 1 < 128) {
      unsigned nn2 = (unsigned)(nstart + (t + 1) * 32) * 2u;
      unsigned bs = (unsigned)(cur ^ 1) * bufB;
#pragma unroll
      for (int j = 0; j < 4; ++j) async_b128(q_l[j] + bs, qg, q_g[j] + nn2);
#pragma unroll
      for (int j = 0; j < 2; ++j) async_b128(k_l[j] + bs, kg, k_g[j] + nn2);
    }
    const unsigned short* lq = &sm[cur][0];
    const unsigned short* lk = &sm[cur][T128US];
    Frag16 A0, A1;
    load_frag_a(A0, lq, m0, lane);
    load_frag_a(A1, lq, m0 + 16, lane);
#pragma unroll
    for (int j = 0; j < 4; ++j) {
      Frag16 B;
      load_frag_b(B, lk, j * 16, lane);
      acc0[j] = wmma_bf16(A0, B, acc0[j]);
      acc1[j] = wmma_bf16(A1, B, acc1[j]);
    }
    wait_async0();
    __syncthreads();
  }

  int lane15 = lane & 15, hf = lane >> 4;
#pragma unroll
  for (int r = 0; r < 8; ++r) {
#pragma unroll
    for (int j = 0; j < 4; ++j) {
      int cA = c0 + m0 + r + 8 * hf;
      qkp[(((ksp * 8 + bat) * CCH + cA) << 8) + d0 + j * 16 + lane15] = acc0[j][r];
      qkp[(((ksp * 8 + bat) * CCH + cA + 16) << 8) + d0 + j * 16 + lane15] = acc1[j][r];
    }
  }
}

// ---------- kernel 4: softmax over d (sums the 4 split-K partials) ----------
__global__ __launch_bounds__(256) void softmax_kernel(
    const float* __restrict__ qkp, unsigned short* __restrict__ attn) {
  __shared__ float red[256];
  int c = blockIdx.x, bat = blockIdx.y, d = threadIdx.x;
  float s = 0.0f;
#pragma unroll
  for (int p = 0; p < 4; ++p)
    s += qkp[(((p * 8 + bat) * CCH + c) << 8) + d];
  red[d] = s;
  __syncthreads();
  for (int off = 128; off > 0; off >>= 1) {
    if (d < off) red[d] = fmaxf(red[d], red[d + off]);
    __syncthreads();
  }
  float mx = red[0];
  __syncthreads();
  float e = __expf(s - mx);
  red[d] = e;
  __syncthreads();
  for (int off = 128; off > 0; off >>= 1) {
    if (d < off) red[d] += red[d + off];
    __syncthreads();
  }
  attn[((bat * CCH + c) << 8) + d] = f32_to_bf16(e / red[0]);
}

// ---------- kernel 5: out = attn @ v, 128x64 tile, async double-buffered ----------
__global__ __launch_bounds__(128) void av_kernel(
    const unsigned short* __restrict__ attn, const unsigned short* __restrict__ vtg,
    float* __restrict__ out) {
  __shared__ __align__(16) unsigned short sm[2][T128US + T64US];
  int tid = threadIdx.x, lane = tid & 31, wid = tid >> 5;
  int n0 = blockIdx.x * 64;
  int c0 = blockIdx.y * 128;
  int bat = blockIdx.z;
  int m0  = wid * 32;

  unsigned smb = lds_addr_of(&sm[0][0]);
  const unsigned bufB = (T128US + T64US) * 2;
  unsigned a_l[4], a_g[4], b_l[2], b_g[2];
#pragma unroll
  for (int j = 0; j < 4; ++j) {
    int ch = tid + j * 128, r = ch >> 2, qo = ch & 3;
    a_l[j] = smb + row_b(r) + qo * 16;
    a_g[j] = (unsigned)((bat * CCH + c0 + r) << 8) * 2u + qo * 16;
  }
#pragma unroll
  for (int j = 0; j < 2; ++j) {
    int ch = tid + j * 128, r = ch >> 2, qo = ch & 3;
    b_l[j] = smb + T128US * 2 + row_b(r) + qo * 16;
    b_g[j] = (unsigned)((bat * HWN + n0 + r) << 8) * 2u + qo * 16;
  }

  v8f acc0[4], acc1[4];
#pragma unroll
  for (int j = 0; j < 4; ++j) { acc0[j] = zero8(); acc1[j] = zero8(); }

  {
#pragma unroll
    for (int j = 0; j < 4; ++j) async_b128(a_l[j], attn, a_g[j]);
#pragma unroll
    for (int j = 0; j < 2; ++j) async_b128(b_l[j], vtg, b_g[j]);
  }
  wait_async0();
  __syncthreads();

  for (int t = 0; t < 8; ++t) {
    int cur = t & 1;
    if (t + 1 < 8) {
      unsigned kd2 = (unsigned)((t + 1) * 32) * 2u;
      unsigned bs = (unsigned)(cur ^ 1) * bufB;
#pragma unroll
      for (int j = 0; j < 4; ++j) async_b128(a_l[j] + bs, attn, a_g[j] + kd2);
#pragma unroll
      for (int j = 0; j < 2; ++j) async_b128(b_l[j] + bs, vtg, b_g[j] + kd2);
    }
    const unsigned short* la = &sm[cur][0];
    const unsigned short* lb = &sm[cur][T128US];
    Frag16 A0, A1;
    load_frag_a(A0, la, m0, lane);
    load_frag_a(A1, la, m0 + 16, lane);
#pragma unroll
    for (int j = 0; j < 4; ++j) {
      Frag16 B;
      load_frag_b(B, lb, j * 16, lane);
      acc0[j] = wmma_bf16(A0, B, acc0[j]);
      acc1[j] = wmma_bf16(A1, B, acc1[j]);
    }
    wait_async0();
    __syncthreads();
  }

  int lane15 = lane & 15, hf = lane >> 4;
#pragma unroll
  for (int r = 0; r < 8; ++r) {
#pragma unroll
    for (int j = 0; j < 4; ++j) {
      int cA = c0 + m0 + r + 8 * hf;
      out[(bat * CCH + cA) * HWN + n0 + j * 16 + lane15] = acc0[j][r];
      out[(bat * CCH + cA + 16) * HWN + n0 + j * 16 + lane15] = acc1[j][r];
    }
  }
}

// ---------- launch ----------
extern "C" void kernel_launch(void* const* d_in, const int* in_sizes, int n_in,
                              void* d_out, int out_size, void* d_ws, size_t ws_size,
                              hipStream_t stream) {
  const float* a  = (const float*)d_in[0];
  const float* b  = (const float*)d_in[1];
  const float* w1 = (const float*)d_in[2];
  const float* b1 = (const float*)d_in[3];
  const float* w2 = (const float*)d_in[4];
  const float* b2 = (const float*)d_in[5];
  const float* w3 = (const float*)d_in[6];
  const float* b3 = (const float*)d_in[7];
  const float* w4 = (const float*)d_in[8];
  const float* b4 = (const float*)d_in[9];

  char* ws = (char*)d_ws;
  const size_t qkvElems = (size_t)8 * CCH * HWN;
  unsigned short* Wbf   = (unsigned short*)ws;                         // 384 KB
  float*          biasf = (float*)(ws + 393216);                       // 3 KB
  unsigned short* qg    = (unsigned short*)(ws + 397312);
  unsigned short* kg    = qg + qkvElems;
  unsigned short* vtg   = kg + qkvElems;
  float*          qkp   = (float*)((char*)vtg + qkvElems * 2);         // 8 MB
  unsigned short* attn  = (unsigned short*)((char*)qkp +
                                            (size_t)4 * 8 * CCH * CCH * 4);

  compose_kernel<<<dim3(768), 256, 0, stream>>>(w1, b1, w2, b2, w3, b3, w4, b4,
                                                Wbf, biasf);
  qkv_kernel<<<dim3(HWN / 64, 2, 8), 256, 0, stream>>>(a, b, Wbf, biasf,
                                                       qg, kg, vtg);
  qk_kernel<<<dim3(8, 4, 8), 128, 0, stream>>>(qg, kg, qkp);
  softmax_kernel<<<dim3(CCH, 8), 256, 0, stream>>>(qkp, attn);
  av_kernel<<<dim3(HWN / 64, 2, 8), 128, 0, stream>>>(attn, vtg, (float*)d_out);
}
